// ChainCRF_38628935860689
// MI455X (gfx1250) — matrix operand order repaired and checked by reading, not compiled, not run on hardware
//
#include <hip/hip_runtime.h>

// ---------------------------------------------------------------------------
// ChainCRF loss on MI455X (gfx1250, wave32, WMMA + async-to-LDS)
// B=32, L=256, D=768, K=51 (padded to 64), N' = 51*64 = 3264 GEMM columns.
// ---------------------------------------------------------------------------

#define BB   32
#define LL   256
#define DD   768
#define KK   51
#define KP   64
#define MM   (BB * LL)        // 8192 rows
#define NP   (KK * KP)        // 3264 padded columns (i*64 + j, i<51)
#define NEG  (-1e30f)

typedef __attribute__((ext_vector_type(16))) __bf16    v16bf;
typedef __attribute__((ext_vector_type(8)))  __bf16    v8bf;
typedef __attribute__((ext_vector_type(16))) _Float16  v16h;
typedef __attribute__((ext_vector_type(8)))  _Float16  v8h;
typedef __attribute__((ext_vector_type(8)))  float     v8f;

// ---- fp32 <-> bf16 split helpers (bit-exact RNE, no __bf16 arithmetic) ----
union BfU { __bf16 b; unsigned short u; };

__device__ inline __bf16 f2bf(float f) {
  unsigned x = __float_as_uint(f);
  unsigned r = x + 0x7fffu + ((x >> 16) & 1u);
  BfU t; t.u = (unsigned short)(r >> 16);
  return t.b;
}
__device__ inline float bf2f(__bf16 b) {
  BfU t; t.b = b;
  return __uint_as_float(((unsigned)t.u) << 16);
}

// ---- CDNA5 async global->LDS copy (ASYNCcnt path, bypasses VGPRs) --------
// VGLOBAL encoding: VDST = LDS byte address (addr[31:0] of shared aperture),
// VADDR = 64-bit global address, saddr = off.  Tracked by ASYNCcnt.
__device__ inline void async_ld16(void* lds, const void* g) {
  unsigned l = (unsigned)(size_t)lds;               // LDS byte address = low 32 bits of pointer
  unsigned long long a = (unsigned long long)(size_t)g;
  asm volatile("global_load_async_to_lds_b128 %0, %1, off"
               :: "v"(l), "v"(a) : "memory");
}
__device__ inline void async_wait0() {
  asm volatile("s_wait_asynccnt 0x0" ::: "memory");
}

// ---- WMMA fragment builders (this code's contiguous k order; A and B match) ----
// A 16x32 (16-bit): lane<16 -> K 0..15 contiguous; lane>=16 -> K 16..31
__device__ inline v16bf fragA_bf(const __bf16* p) {
  v8bf x = *(const v8bf*)p;
  v8bf y = *(const v8bf*)(p + 8);
  v16bf r;
#pragma unroll
  for (int e = 0; e < 8; ++e) { r[e] = x[e]; r[e + 8] = y[e]; }
  return r;
}
// B 32x16 (16-bit): per lane 16 contiguous K values of one column
__device__ inline v16bf fragB_bf(const __bf16* p) {
  v8bf x = *(const v8bf*)p;
  v8bf y = *(const v8bf*)(p + 8);
  v16bf r;
#pragma unroll
  for (int e = 0; e < 8; ++e) { r[e] = x[e]; r[e + 8] = y[e]; }
  return r;
}
__device__ inline v16h fragA_h(const _Float16* p) {
  v8h x = *(const v8h*)p;
  v8h y = *(const v8h*)(p + 8);
  v16h r;
#pragma unroll
  for (int e = 0; e < 8; ++e) { r[e] = x[e]; r[e + 8] = y[e]; }
  return r;
}
__device__ inline v16h fragB_h(const _Float16* p) {
  v8h x = *(const v8h*)p;
  v8h y = *(const v8h*)(p + 8);
  v16h r;
#pragma unroll
  for (int e = 0; e < 8; ++e) { r[e] = x[e]; r[e + 8] = y[e]; }
  return r;
}

// ---------------------------------------------------------------------------
// Kernel 0: fill padded energy buffer with -1e30, zero output scalar
// ---------------------------------------------------------------------------
__global__ void crf_init(float* __restrict__ E, float* __restrict__ out) {
  size_t idx = (size_t)blockIdx.x * blockDim.x + threadIdx.x;
  const size_t n4 = (size_t)MM * (KP * KP) / 4;
  float4 v = make_float4(NEG, NEG, NEG, NEG);
  for (size_t i = idx; i < n4; i += (size_t)gridDim.x * blockDim.x)
    ((float4*)E)[i] = v;
  if (idx == 0) out[0] = 0.0f;
}

// ---------------------------------------------------------------------------
// Kernel 1a: split x into bf16 hi/lo
// ---------------------------------------------------------------------------
__global__ void crf_prep_x(const float* __restrict__ x,
                           __bf16* __restrict__ xh, __bf16* __restrict__ xl) {
  int idx = blockIdx.x * blockDim.x + threadIdx.x;
  if (idx >= MM * DD) return;
  float v = x[idx];
  __bf16 h = f2bf(v);
  xh[idx] = h;
  xl[idx] = f2bf(v - bf2f(h));
}

// ---------------------------------------------------------------------------
// Kernel 1b: build combined padded weights Wc[c'=i*64+j] = trans_W[i*51+j] +
//            state_W[j] (zero in pad columns), split hi/lo; combined bias.
// ---------------------------------------------------------------------------
__global__ void crf_prep_w(const float* __restrict__ tW, const float* __restrict__ tb,
                           const float* __restrict__ sW, const float* __restrict__ sb,
                           __bf16* __restrict__ wh, __bf16* __restrict__ wl,
                           float* __restrict__ bc) {
  int idx = blockIdx.x * blockDim.x + threadIdx.x;
  if (idx >= NP * DD) return;
  int cp = idx / DD, k = idx - cp * DD;
  int j = cp & 63, i = cp >> 6;
  float v = 0.0f;
  if (j < KK) v = tW[(i * KK + j) * DD + k] + sW[j * DD + k];
  __bf16 h = f2bf(v);
  wh[idx] = h;
  wl[idx] = f2bf(v - bf2f(h));
  if (k == 0) bc[cp] = (j < KK) ? (tb[i * KK + j] + sb[j]) : 0.0f;
}

// ---------------------------------------------------------------------------
// Kernel 2: energy GEMM.  D = Xsplit @ Wc^T via bf16 WMMA (hi*hi+hi*lo+lo*hi).
// Block tile 128(M) x 64(N), 8 waves of 32x32.  Staging via async-to-LDS.
// Epilogue: +bias, *mask, identity substitution for masked t>0, padded store.
// ---------------------------------------------------------------------------
__global__ __launch_bounds__(256) void crf_energy_gemm(
    const __bf16* __restrict__ XH, const __bf16* __restrict__ XL,
    const __bf16* __restrict__ WH, const __bf16* __restrict__ WL,
    const float* __restrict__ BC, const float* __restrict__ mask,
    float* __restrict__ E) {
  __shared__ __bf16 As[2][128][32];   // [hi/lo][m][k]
  __shared__ __bf16 Bs[2][64][32];    // [hi/lo][n][k]

  const int tid  = threadIdx.x;
  const int lane = tid & 31;
  const int w    = tid >> 5;
  const int bm   = blockIdx.y * 128;
  const int bn   = blockIdx.x * 64;
  const int wm   = (w & 3) * 32;
  const int wn   = (w >> 2) * 32;

  v8f acc[2][2] = {};

  for (int k0 = 0; k0 < DD; k0 += 32) {
    __syncthreads();                  // previous LDS contents fully consumed
    // stage A: 128x32 bf16 (hi and lo) via async DMA, 16B per transfer
#pragma unroll
    for (int c = tid; c < 512; c += 256) {
      int r = c >> 2, kk = (c & 3) << 3;
      int g = (bm + r) * DD + k0 + kk;
      async_ld16(&As[0][r][kk], &XH[g]);
      async_ld16(&As[1][r][kk], &XL[g]);
    }
    // stage B: 64x32 bf16 (hi and lo)
    {
      int r = tid >> 2, kk = (tid & 3) << 3;
      int g = (bn + r) * DD + k0 + kk;
      async_ld16(&Bs[0][r][kk], &WH[g]);
      async_ld16(&Bs[1][r][kk], &WL[g]);
    }
    async_wait0();                    // ASYNCcnt == 0 for this wave
    __syncthreads();                  // all waves' tiles visible

    const int arow = wm + (lane & 15);
    const int ac   = (lane < 16) ? 0 : 16;
    const int bcol = wn + (lane & 15);
    const int bkb  = (lane < 16) ? 0 : 16;

    v16bf ah[2], al[2], bh[2], bl[2];
#pragma unroll
    for (int mi = 0; mi < 2; ++mi) {
      ah[mi] = fragA_bf(&As[0][arow + mi * 16][ac]);
      al[mi] = fragA_bf(&As[1][arow + mi * 16][ac]);
    }
#pragma unroll
    for (int ni = 0; ni < 2; ++ni) {
      bh[ni] = fragB_bf(&Bs[0][bcol + ni * 16][bkb]);
      bl[ni] = fragB_bf(&Bs[1][bcol + ni * 16][bkb]);
    }
#pragma unroll
    for (int mi = 0; mi < 2; ++mi)
#pragma unroll
      for (int ni = 0; ni < 2; ++ni) {
        acc[mi][ni] = __builtin_amdgcn_wmma_f32_16x16x32_bf16(
            false, ah[mi], false, bh[ni], (short)0, acc[mi][ni], false, false);
        acc[mi][ni] = __builtin_amdgcn_wmma_f32_16x16x32_bf16(
            false, ah[mi], false, bl[ni], (short)0, acc[mi][ni], false, false);
        acc[mi][ni] = __builtin_amdgcn_wmma_f32_16x16x32_bf16(
            false, al[mi], false, bh[ni], (short)0, acc[mi][ni], false, false);
      }
  }

  // Epilogue. C layout: lane<16 -> N=lane, M=r; lane>=16 -> N=lane-16, M=8+r.
  const int n_loc = lane & 15;
  const int m_add = (lane < 16) ? 0 : 8;
#pragma unroll
  for (int mi = 0; mi < 2; ++mi)
#pragma unroll
    for (int ni = 0; ni < 2; ++ni) {
      int cp = bn + wn + ni * 16 + n_loc;        // padded column i*64+j
      int j = cp & 63, i = cp >> 6;
      if (j >= KK) continue;                      // leave pad at -1e30
      float bias = BC[cp];
#pragma unroll
      for (int r = 0; r < 8; ++r) {
        int row = bm + wm + mi * 16 + r + m_add;  // bl = b*256 + t
        float m = mask[row];
        int t = row & (LL - 1);
        float v = (acc[mi][ni][r] + bias) * m;
        if (t > 0 && m == 0.0f) v = (i == j) ? 0.0f : NEG;  // semiring identity
        E[(size_t)row * (KP * KP) + cp] = v;
      }
    }
}

// ---------------------------------------------------------------------------
// Kernel 3: target-path energy (runs before tree overwrites E).
// ---------------------------------------------------------------------------
__global__ __launch_bounds__(256) void crf_gather(
    const float* __restrict__ E, const float* __restrict__ mask,
    const int* __restrict__ tgt, float* __restrict__ te) {
  __shared__ float sm[LL];
  __shared__ int   st[LL];
  __shared__ int   pb[LL];
  __shared__ float red[LL];
  int b = blockIdx.x, tid = threadIdx.x;
  sm[tid] = mask[b * LL + tid];
  st[tid] = tgt[b * LL + tid];
  __syncthreads();
  if (tid == 0) {                       // prev-label scan (cheap, ALU only)
    int prev = (sm[0] != 0.0f) ? st[0] : (KK - 1);
    for (int t = 1; t < LL; ++t) { pb[t] = prev; if (sm[t] != 0.0f) prev = st[t]; }
  }
  __syncthreads();
  float c;
  if (tid == 0)
    c = E[(size_t)(b * LL) * (KP * KP) + (KK - 1) * KP + st[0]];
  else
    c = (sm[tid] != 0.0f)
            ? E[(size_t)(b * LL + tid) * (KP * KP) + pb[tid] * KP + st[tid]]
            : 0.0f;
  red[tid] = c;
  __syncthreads();
  for (int s = 128; s > 0; s >>= 1) {
    if (tid < s) red[tid] += red[tid + s];
    __syncthreads();
  }
  if (tid == 0) te[b] = red[0];
}

// ---------------------------------------------------------------------------
// Kernel 4: one tree level of log-semiring matmul: C = A (x) B, 64x64x64.
// Stabilized: C = amax_i + bmax_k + log( exp(A-amax) @ exp(B-bmax) ), inner
// product via f16 WMMA.  In-place tree: result written to A's slot.
// ---------------------------------------------------------------------------
__global__ __launch_bounds__(256) void crf_combine(
    float* __restrict__ E, int pairsPerB, int stride) {
  __shared__ float    Af[64][64];
  __shared__ float    Bf[64][64];
  __shared__ _Float16 P[64][64];    // exp(A[i][j]-amax[i])
  __shared__ _Float16 Qt[64][64];   // exp(B[j][n]-bmax[n]) stored [n][j]
  __shared__ float    amax[64], bmax[64];

  int blk = blockIdx.x;
  int b = blk / pairsPerB, p = blk - b * pairsPerB;
  size_t baseA = (size_t)(b * LL + p * 2 * stride) * (KP * KP);
  size_t baseB = baseA + (size_t)stride * (KP * KP);
  int tid = threadIdx.x;

  // stage both 16 KB tiles via async DMA (16B granules)
  for (int i4 = tid; i4 < 1024; i4 += 256) {
    async_ld16(&Af[0][0] + i4 * 4, E + baseA + i4 * 4);
    async_ld16(&Bf[0][0] + i4 * 4, E + baseB + i4 * 4);
  }
  async_wait0();
  __syncthreads();
  if (tid < 64) {
    float m1 = -3.4e38f, m2 = -3.4e38f;
    for (int jj = 0; jj < 64; ++jj) {
      m1 = fmaxf(m1, Af[tid][jj]);
      m2 = fmaxf(m2, Bf[jj][tid]);
    }
    amax[tid] = m1;
    bmax[tid] = m2;
  }
  __syncthreads();
  for (int e = tid; e < 4096; e += 256) {
    int i = e >> 6, jj = e & 63;
    P[i][jj]  = (_Float16)__expf(Af[i][jj] - amax[i]);
    Qt[i][jj] = (_Float16)__expf(Bf[jj][i] - bmax[i]);
  }
  __syncthreads();

  int lane = tid & 31, w = tid >> 5;
  int wm = (w & 3) * 16, wn = (w >> 2) * 32;   // wave tile 16x32
  v8f acc[2] = {};
#pragma unroll
  for (int ks = 0; ks < 64; ks += 32) {
    v16h a = fragA_h(&P[wm + (lane & 15)][ks + ((lane < 16) ? 0 : 16)]);
#pragma unroll
    for (int ni = 0; ni < 2; ++ni) {
      v16h bb = fragB_h(&Qt[wn + ni * 16 + (lane & 15)][ks + ((lane < 16) ? 0 : 16)]);
      acc[ni] = __builtin_amdgcn_wmma_f32_16x16x32_f16(
          false, a, false, bb, (short)0, acc[ni], false, false);
    }
  }
  const int m_add = (lane < 16) ? 0 : 8;
#pragma unroll
  for (int ni = 0; ni < 2; ++ni) {
    int k = wn + ni * 16 + (lane & 15);
    float bmk = bmax[k];
#pragma unroll
    for (int r = 0; r < 8; ++r) {
      int i = wm + r + m_add;
      float s = acc[ni][r];
      E[baseA + i * 64 + k] = amax[i] + bmk + __logf(fmaxf(s, 1e-30f));
    }
  }
}

// ---------------------------------------------------------------------------
// Kernel 5: loss = mean_b( logsumexp_k(product_row50[k]) - te[b] )
// ---------------------------------------------------------------------------
__global__ void crf_finalize(const float* __restrict__ E,
                             const float* __restrict__ te,
                             float* __restrict__ out) {
  __shared__ float sv[64];
  int b = blockIdx.x, tid = threadIdx.x;  // 64 threads
  float v = (tid < KK)
                ? E[(size_t)(b * LL) * (KP * KP) + (KK - 1) * KP + tid]
                : NEG;
  sv[tid] = v;
  __syncthreads();
  for (int s = 32; s > 0; s >>= 1) {
    if (tid < s) sv[tid] = fmaxf(sv[tid], sv[tid + s]);
    __syncthreads();
  }
  float mx = sv[0];
  __syncthreads();
  sv[tid] = __expf(v - mx);
  __syncthreads();
  for (int s = 32; s > 0; s >>= 1) {
    if (tid < s) sv[tid] += sv[tid + s];
    __syncthreads();
  }
  if (tid == 0) {
    float lse = mx + __logf(sv[0]);
    atomicAdd(out, (lse - te[b]) * (1.0f / (float)BB));
  }
}

// ---------------------------------------------------------------------------
// Host-side launch
// ---------------------------------------------------------------------------
extern "C" void kernel_launch(void* const* d_in, const int* in_sizes, int n_in,
                              void* d_out, int out_size, void* d_ws, size_t ws_size,
                              hipStream_t stream) {
  const float* x    = (const float*)d_in[0];
  const float* mask = (const float*)d_in[1];
  const int*   tgt  = (const int*)d_in[2];
  const float* sW   = (const float*)d_in[3];
  const float* sb   = (const float*)d_in[4];
  const float* tW   = (const float*)d_in[5];
  const float* tb   = (const float*)d_in[6];
  float* out = (float*)d_out;

  // workspace layout (bytes)
  const size_t SZ_E  = (size_t)MM * KP * KP * 4;        // 134 MB padded energy
  const size_t SZ_X  = (size_t)MM * DD * 2;             // bf16 split of x
  const size_t SZ_W  = (size_t)NP * DD * 2;             // bf16 split of Wc
  char* ws = (char*)d_ws;
  float*  E  = (float*)ws;
  __bf16* XH = (__bf16*)(ws + SZ_E);
  __bf16* XL = (__bf16*)(ws + SZ_E + SZ_X);
  __bf16* WH = (__bf16*)(ws + SZ_E + 2 * SZ_X);
  __bf16* WL = (__bf16*)(ws + SZ_E + 2 * SZ_X + SZ_W);
  float*  BC = (float*)(ws + SZ_E + 2 * SZ_X + 2 * SZ_W);
  float*  TE = (float*)(ws + SZ_E + 2 * SZ_X + 2 * SZ_W + (size_t)NP * 4);

  crf_init<<<2048, 256, 0, stream>>>(E, out);
  crf_prep_x<<<(MM * DD + 255) / 256, 256, 0, stream>>>(x, XH, XL);
  crf_prep_w<<<(NP * DD + 255) / 256, 256, 0, stream>>>(tW, tb, sW, sb, WH, WL, BC);

  dim3 gg(NP / 64, MM / 128);   // 51 x 64 blocks
  crf_energy_gemm<<<gg, 256, 0, stream>>>(XH, XL, WH, WL, BC, mask, E);

  crf_gather<<<BB, 256, 0, stream>>>(E, mask, tgt, TE);

  for (int lvl = 0; lvl < 8; ++lvl) {
    int pairs = 128 >> lvl;
    crf_combine<<<BB * pairs, 256, 0, stream>>>(E, pairs, 1 << lvl);
  }

  crf_finalize<<<BB, 64, 0, stream>>>(E, TE, out);
}
